// DiffractionIntegration_6906307412185
// MI455X (gfx1250) — matrix operand order repaired and checked by reading, not compiled, not run on hardware
//
#include <hip/hip_runtime.h>
#include <hip/hip_bf16.h>

// ---------------------------------------------------------------------------
// Types for WMMA fragments (CDNA5 / gfx1250, wave32)
// ---------------------------------------------------------------------------
typedef __attribute__((ext_vector_type(16))) _Float16 v16h;
typedef __attribute__((ext_vector_type(8)))  float    v8f;

union AF { v16h h; uint4 u[2]; };

#define TILE  64     // node rows per workgroup (200000 = 3125 * 64 exactly)
#define MT    (TILE / 16)   // m-tiles per workgroup = 4
#define PA    264    // f16 LDS pitch for 256-wide activations (mult of 8)
#define PA3   136    // f16 LDS pitch for 128-wide activations (mult of 8)
#define PACC  312    // f32 LDS pitch for accumulator rows (mult of 8)
#define NHKL  300
#define NHKLP 304    // padded to multiple of 16
#define BATCH 256
#define NODE_F 256
#define GRAPH_F 512

// Dynamic LDS layout (bytes), all 16B aligned:
//   sA    : TILE*PA*2   = 33792
//   sAcc  : TILE*PACC*4 = 79872   -> offset 33792
//   sPos  : TILE*3*4    = 768     -> offset 113664
//   sBatch: TILE*4      = 256     -> offset 114432
//   sHkl  : NHKLP*3*4   = 3648    -> offset 114688
//   total               = 118336
#define SMEM_BYTES 118336

__device__ __forceinline__ float wred32(float v) {
  #pragma unroll
  for (int m = 16; m; m >>= 1) v += __shfl_xor(v, m, 32);
  return v;
}

// ---------------------------------------------------------------------------
// Kernel 0: weight prep — f32 [K,N] -> f16 transposed [N][K] (B-matrix order)
// ---------------------------------------------------------------------------
__global__ void prep_weights(const float* __restrict__ w1,
                             const float* __restrict__ w2,
                             const float* __restrict__ w3,
                             _Float16* __restrict__ w1T,
                             _Float16* __restrict__ w2T,
                             _Float16* __restrict__ w3T) {
  int i = blockIdx.x * 256 + threadIdx.x;   // launched with 65536 threads
  if (i < 256 * 256) {
    int n = i >> 8, k = i & 255;
    w1T[i] = (_Float16)w1[k * 256 + n];
  }
  if (i < 128 * 256) {
    int n = i >> 8, k = i & 255;
    w2T[i] = (_Float16)w2[k * 128 + n];
  }
  if (i < NHKLP * 128) {
    int n = i >> 7, k = i & 127;
    w3T[i] = (_Float16)((n < NHKL) ? w3[k * NHKL + n] : 0.f);
  }
}

// ---------------------------------------------------------------------------
// Kernel 1: zero the segment accumulators (real+imag are contiguous)
// ---------------------------------------------------------------------------
__global__ void zero_accum(float* __restrict__ p, int n) {
  int i = blockIdx.x * 256 + threadIdx.x;
  if (i < n) p[i] = 0.f;
}

// ---------------------------------------------------------------------------
// Kernel 2: fused per-node pipeline (64 nodes / workgroup).
// Each wave owns n-tiles; one B (weight) fragment is reused across all 4
// m-tiles, so each weight element is fetched exactly once per workgroup.
// ---------------------------------------------------------------------------
__global__ __launch_bounds__(256)
void fused_node_kernel(const float* __restrict__ X,
                       const float* __restrict__ pos,
                       const int*   __restrict__ batch,
                       const float* __restrict__ hkl,
                       const _Float16* __restrict__ w1T, const float* __restrict__ b1,
                       const float* __restrict__ g1,     const float* __restrict__ be1,
                       const _Float16* __restrict__ w2T, const float* __restrict__ b2,
                       const float* __restrict__ g2,     const float* __restrict__ be2,
                       const _Float16* __restrict__ w3T, const float* __restrict__ b3,
                       float* __restrict__ realW, float* __restrict__ imagW) {
  extern __shared__ __align__(16) char smem[];
  _Float16* sA    = (_Float16*)(smem);
  float*    sAcc  = (float*)(smem + 33792);
  float*    sPos  = (float*)(smem + 113664);
  int*      sBatch = (int*)(smem + 114432);
  float*    sHkl  = (float*)(smem + 114688);

  const int tid  = threadIdx.x;
  const int lane = tid & 31;
  const int wave = tid >> 5;
  const int base = blockIdx.x * TILE;

  // ---- stage 0: stage inputs into LDS (X converted f32 -> f16, b128 loads) --
  {
    const float4* X4 = (const float4*)X;   // 64 float4 per row
    for (int i = tid; i < TILE * 64; i += 256) {
      int r = i >> 6, c4 = i & 63;
      float4 v = X4[(size_t)(base + r) * 64 + c4];
      _Float16* d = sA + r * PA + c4 * 4;
      d[0] = (_Float16)v.x; d[1] = (_Float16)v.y;
      d[2] = (_Float16)v.z; d[3] = (_Float16)v.w;
    }
  }
  for (int i = tid; i < TILE * 3; i += 256) sPos[i] = pos[(size_t)base * 3 + i];
  if (tid < TILE) sBatch[tid] = batch[base + tid];
  for (int i = tid; i < NHKLP * 3; i += 256) sHkl[i] = (i < NHKL * 3) ? hkl[i] : 0.f;
  __syncthreads();

  const int mSub  = lane & 15;       // row/col within 16
  const int khalf = lane >> 4;       // A: +8 K offset; B: +8 K offset
  const int mOff  = (lane >> 4) * 8; // C/D row offset (f32 C layout)

  // ---- stage 1: H1 = X @ W1 + b1  (M=64, N=256, K=256) ----
  for (int nt = wave; nt < 16; nt += 8) {
    const int n = nt * 16 + mSub;
    v8f acc[MT];
    #pragma unroll
    for (int mt = 0; mt < MT; ++mt) acc[mt] = (v8f){};
    for (int kt = 0; kt < 8; ++kt) {
      const int kb = kt * 32;
      AF b;
      const uint4* pb = (const uint4*)(w1T + n * 256 + kb + khalf * 8);
      b.u[0] = pb[0]; b.u[1] = pb[2];
      #pragma unroll
      for (int mt = 0; mt < MT; ++mt) {
        AF a;
        const _Float16* ap = sA + (mt * 16 + mSub) * PA + kb + khalf * 8;
        a.u[0] = *(const uint4*)(ap);
        a.u[1] = *(const uint4*)(ap + 16);
        acc[mt] = __builtin_amdgcn_wmma_f32_16x16x32_f16(false, a.h, false, b.h,
                                                         (short)0, acc[mt], false, false);
      }
    }
    const float bn = b1[n];
    #pragma unroll
    for (int mt = 0; mt < MT; ++mt)
      #pragma unroll
      for (int i = 0; i < 8; ++i)
        sAcc[(mt * 16 + mOff + i) * PACC + n] = acc[mt][i] + bn;
  }
  __syncthreads();

  // ---- stage 1b: LayerNorm(256) + SiLU -> f16 into sA ----
  for (int rr = 0; rr < TILE / 8; ++rr) {
    const int row = wave * (TILE / 8) + rr;
    float x[8], s = 0.f;
    #pragma unroll
    for (int j = 0; j < 8; ++j) { x[j] = sAcc[row * PACC + lane + 32 * j]; s += x[j]; }
    s = wred32(s);
    const float mean = s * (1.f / 256.f);
    float v = 0.f;
    #pragma unroll
    for (int j = 0; j < 8; ++j) { float d = x[j] - mean; v += d * d; }
    v = wred32(v);
    const float rstd = rsqrtf(v * (1.f / 256.f) + 1e-5f);
    #pragma unroll
    for (int j = 0; j < 8; ++j) {
      const int c = lane + 32 * j;
      float y = (x[j] - mean) * rstd * g1[c] + be1[c];
      sA[row * PA + c] = (_Float16)(y / (1.f + __expf(-y)));
    }
  }
  __syncthreads();

  // ---- stage 2: H2 = H1 @ W2 + b2  (M=64, N=128, K=256) ----
  {
    const int nt = wave;               // 8 n-tiles over 8 waves
    const int n = nt * 16 + mSub;
    v8f acc[MT];
    #pragma unroll
    for (int mt = 0; mt < MT; ++mt) acc[mt] = (v8f){};
    for (int kt = 0; kt < 8; ++kt) {
      const int kb = kt * 32;
      AF b;
      const uint4* pb = (const uint4*)(w2T + n * 256 + kb + khalf * 8);
      b.u[0] = pb[0]; b.u[1] = pb[2];
      #pragma unroll
      for (int mt = 0; mt < MT; ++mt) {
        AF a;
        const _Float16* ap = sA + (mt * 16 + mSub) * PA + kb + khalf * 8;
        a.u[0] = *(const uint4*)(ap);
        a.u[1] = *(const uint4*)(ap + 16);
        acc[mt] = __builtin_amdgcn_wmma_f32_16x16x32_f16(false, a.h, false, b.h,
                                                         (short)0, acc[mt], false, false);
      }
    }
    const float bn = b2[n];
    #pragma unroll
    for (int mt = 0; mt < MT; ++mt)
      #pragma unroll
      for (int i = 0; i < 8; ++i)
        sAcc[(mt * 16 + mOff + i) * PACC + n] = acc[mt][i] + bn;
  }
  __syncthreads();

  // ---- stage 2b: LayerNorm(128) + SiLU -> f16 into sA (pitch PA3) ----
  for (int rr = 0; rr < TILE / 8; ++rr) {
    const int row = wave * (TILE / 8) + rr;
    float x[4], s = 0.f;
    #pragma unroll
    for (int j = 0; j < 4; ++j) { x[j] = sAcc[row * PACC + lane + 32 * j]; s += x[j]; }
    s = wred32(s);
    const float mean = s * (1.f / 128.f);
    float v = 0.f;
    #pragma unroll
    for (int j = 0; j < 4; ++j) { float d = x[j] - mean; v += d * d; }
    v = wred32(v);
    const float rstd = rsqrtf(v * (1.f / 128.f) + 1e-5f);
    #pragma unroll
    for (int j = 0; j < 4; ++j) {
      const int c = lane + 32 * j;
      float y = (x[j] - mean) * rstd * g2[c] + be2[c];
      sA[row * PA3 + c] = (_Float16)(y / (1.f + __expf(-y)));
    }
  }
  __syncthreads();

  // ---- stage 3: FF = H2 @ W3 + b3  (M=64, N=304, K=128) ----
  for (int nt = wave; nt < NHKLP / 16; nt += 8) {   // 19 n-tiles
    const int n = nt * 16 + mSub;
    v8f acc[MT];
    #pragma unroll
    for (int mt = 0; mt < MT; ++mt) acc[mt] = (v8f){};
    for (int kt = 0; kt < 4; ++kt) {
      const int kb = kt * 32;
      AF b;
      const uint4* pb = (const uint4*)(w3T + n * 128 + kb + khalf * 8);
      b.u[0] = pb[0]; b.u[1] = pb[2];
      #pragma unroll
      for (int mt = 0; mt < MT; ++mt) {
        AF a;
        const _Float16* ap = sA + (mt * 16 + mSub) * PA3 + kb + khalf * 8;
        a.u[0] = *(const uint4*)(ap);
        a.u[1] = *(const uint4*)(ap + 16);
        acc[mt] = __builtin_amdgcn_wmma_f32_16x16x32_f16(false, a.h, false, b.h,
                                                         (short)0, acc[mt], false, false);
      }
    }
    const float bn = (n < NHKL) ? b3[n] : 0.f;
    #pragma unroll
    for (int mt = 0; mt < MT; ++mt)
      #pragma unroll
      for (int i = 0; i < 8; ++i)
        sAcc[(mt * 16 + mOff + i) * PACC + n] = acc[mt][i] + bn;
  }
  __syncthreads();

  // ---- stage 4: phase, sincos, segment-run reduction + atomics ----
  const float TWO_PI = 6.28318530717958647692f;
  for (int cc = 0; cc < 2; ++cc) {
    const int col = tid + 256 * cc;
    if (col >= NHKLP) break;           // padded cols 300..303 contribute 0 anyway
    const float hx = sHkl[col * 3 + 0] * TWO_PI;
    const float hy = sHkl[col * 3 + 1] * TWO_PI;
    const float hz = sHkl[col * 3 + 2] * TWO_PI;
    float re = 0.f, im = 0.f;
    int cur = sBatch[0];
    for (int r = 0; r < TILE; ++r) {
      const int seg = sBatch[r];
      if (seg != cur) {                // batch is sorted: few flushes per tile
        atomicAdd(&realW[cur * NHKLP + col], re);
        atomicAdd(&imagW[cur * NHKLP + col], im);
        re = 0.f; im = 0.f; cur = seg;
      }
      const float ph = sPos[r * 3] * hx + sPos[r * 3 + 1] * hy + sPos[r * 3 + 2] * hz;
      float sn, cs;
      __sincosf(ph, &sn, &cs);
      const float f = sAcc[r * PACC + col];
      re += f * cs;
      im += f * sn;
    }
    atomicAdd(&realW[cur * NHKLP + col], re);
    atomicAdd(&imagW[cur * NHKLP + col], im);
  }
}

// ---------------------------------------------------------------------------
// Kernel 3: batch-side network (tiny: ~0.8 GFLOP). One block per batch row.
// ---------------------------------------------------------------------------
__device__ __forceinline__ float blockSum(float v, float* red) {
  v = wred32(v);
  __syncthreads();                       // protect red reuse across calls
  if ((threadIdx.x & 31) == 0) red[threadIdx.x >> 5] = v;
  __syncthreads();
  float t = 0.f;
  #pragma unroll
  for (int i = 0; i < 8; ++i) t += red[i];
  return t;
}

__device__ __forceinline__ void ln_silu(float* a, int L, const float* g,
                                        const float* be, float* red) {
  float s = 0.f;
  for (int i = threadIdx.x; i < L; i += 256) s += a[i];
  const float mean = blockSum(s, red) / (float)L;
  float v = 0.f;
  for (int i = threadIdx.x; i < L; i += 256) { float d = a[i] - mean; v += d * d; }
  const float rstd = rsqrtf(blockSum(v, red) / (float)L + 1e-5f);
  for (int i = threadIdx.x; i < L; i += 256) {
    float y = (a[i] - mean) * rstd * g[i] + be[i];
    a[i] = y / (1.f + __expf(-y));
  }
  __syncthreads();
}

__global__ __launch_bounds__(256)
void batch_net(const float* __restrict__ realW, const float* __restrict__ imagW,
               const float* __restrict__ graph,
               const float* __restrict__ w1, const float* __restrict__ b1,
               const float* __restrict__ g1, const float* __restrict__ be1,
               const float* __restrict__ w2, const float* __restrict__ b2,
               const float* __restrict__ g2, const float* __restrict__ be2,
               const float* __restrict__ w3, const float* __restrict__ b3,
               const float* __restrict__ fw1, const float* __restrict__ fb1,
               const float* __restrict__ fg,  const float* __restrict__ fbe,
               const float* __restrict__ fw2, const float* __restrict__ fb2,
               float* __restrict__ out) {
  __shared__ float sf[2 * NHKL];   // 600: interleaved (re, im)
  __shared__ float sg[GRAPH_F];
  __shared__ float t1[512];
  __shared__ float t2[256];
  __shared__ float t3[512];
  __shared__ float red[8];

  const int b = blockIdx.x, tid = threadIdx.x;

  for (int j = tid; j < NHKL; j += 256) {
    sf[2 * j]     = realW[b * NHKLP + j];
    sf[2 * j + 1] = imagW[b * NHKLP + j];
  }
  for (int j = tid; j < GRAPH_F; j += 256) sg[j] = graph[b * GRAPH_F + j];
  __syncthreads();

  // dn layer 1: [600] -> [512]
  for (int o = tid; o < 512; o += 256) {
    float s = b1[o];
    for (int k = 0; k < 2 * NHKL; ++k) s += sf[k] * w1[k * 512 + o];
    t1[o] = s;
  }
  __syncthreads();
  ln_silu(t1, 512, g1, be1, red);

  // dn layer 2: [512] -> [256]
  {
    const int o = tid;
    float s = b2[o];
    for (int k = 0; k < 512; ++k) s += t1[k] * w2[k * 256 + o];
    t2[o] = s;
  }
  __syncthreads();
  ln_silu(t2, 256, g2, be2, red);

  // dn layer 3: [256] -> [512] (reuse t1)
  for (int o = tid; o < 512; o += 256) {
    float s = b3[o];
    for (int k = 0; k < 256; ++k) s += t2[k] * w3[k * 512 + o];
    t1[o] = s;
  }
  __syncthreads();

  // fusion layer 1: comb = [graph | d3] (1024) -> [512]
  for (int o = tid; o < 512; o += 256) {
    float s = fb1[o];
    for (int k = 0; k < 512; ++k) s += sg[k] * fw1[k * 512 + o];
    for (int k = 0; k < 512; ++k) s += t1[k] * fw1[(512 + k) * 512 + o];
    t3[o] = s;
  }
  __syncthreads();
  ln_silu(t3, 512, fg, fbe, red);

  // fusion layer 2 + residual
  for (int o = tid; o < 512; o += 256) {
    float s = fb2[o];
    for (int k = 0; k < 512; ++k) s += t3[k] * fw2[k * 512 + o];
    out[(size_t)b * GRAPH_F + o] = s + sg[o];
  }
}

// ---------------------------------------------------------------------------
// Host launcher
// ---------------------------------------------------------------------------
extern "C" void kernel_launch(void* const* d_in, const int* in_sizes, int n_in,
                              void* d_out, int out_size, void* d_ws, size_t ws_size,
                              hipStream_t stream) {
  const float* graph = (const float*)d_in[0];
  const float* nodef = (const float*)d_in[1];
  const float* pos   = (const float*)d_in[2];
  const int*   batch = (const int*)d_in[3];
  const float* hkl   = (const float*)d_in[4];
  const float* ff_w1 = (const float*)d_in[5];
  const float* ff_b1 = (const float*)d_in[6];
  const float* ln1g  = (const float*)d_in[7];
  const float* ln1b  = (const float*)d_in[8];
  const float* ff_w2 = (const float*)d_in[9];
  const float* ff_b2 = (const float*)d_in[10];
  const float* ln2g  = (const float*)d_in[11];
  const float* ln2b  = (const float*)d_in[12];
  const float* ff_w3 = (const float*)d_in[13];
  const float* ff_b3 = (const float*)d_in[14];
  const float* dn_w1 = (const float*)d_in[15];
  const float* dn_b1 = (const float*)d_in[16];
  const float* dng1  = (const float*)d_in[17];
  const float* dnb1  = (const float*)d_in[18];
  const float* dn_w2 = (const float*)d_in[19];
  const float* dn_b2 = (const float*)d_in[20];
  const float* dng2  = (const float*)d_in[21];
  const float* dnb2  = (const float*)d_in[22];
  const float* dn_w3 = (const float*)d_in[23];
  const float* dn_b3 = (const float*)d_in[24];
  const float* fn_w1 = (const float*)d_in[25];
  const float* fn_b1 = (const float*)d_in[26];
  const float* fng   = (const float*)d_in[27];
  const float* fnb   = (const float*)d_in[28];
  const float* fn_w2 = (const float*)d_in[29];
  const float* fn_b2 = (const float*)d_in[30];

  char* ws = (char*)d_ws;
  const size_t accElems = (size_t)BATCH * NHKLP;           // 77824 floats each
  float*    realW = (float*)(ws + 0);
  float*    imagW = (float*)(ws + accElems * 4);           // 311296
  _Float16* w1T   = (_Float16*)(ws + 2 * accElems * 4);    // 622592
  _Float16* w2T   = (_Float16*)(ws + 2 * accElems * 4 + 256 * 256 * 2);       // 753664
  _Float16* w3T   = (_Float16*)(ws + 2 * accElems * 4 + 256 * 256 * 2 + 128 * 256 * 2); // 819200

  prep_weights<<<256, 256, 0, stream>>>(ff_w1, ff_w2, ff_w3, w1T, w2T, w3T);

  const int zn = 2 * (int)accElems;                         // 155648
  zero_accum<<<(zn + 255) / 256, 256, 0, stream>>>(realW, zn);

  fused_node_kernel<<<200000 / TILE, 256, SMEM_BYTES, stream>>>(
      nodef, pos, batch, hkl,
      w1T, ff_b1, ln1g, ln1b,
      w2T, ff_b2, ln2g, ln2b,
      w3T, ff_b3,
      realW, imagW);

  batch_net<<<BATCH, 256, 0, stream>>>(
      realW, imagW, graph,
      dn_w1, dn_b1, dng1, dnb1,
      dn_w2, dn_b2, dng2, dnb2,
      dn_w3, dn_b3,
      fn_w1, fn_b1, fng, fnb,
      fn_w2, fn_b2,
      (float*)d_out);
}
